// SRU_TPU_38654705664626
// MI455X (gfx1250) — matrix operand order repaired and checked by reading, not compiled, not run on hardware
//
#include <hip/hip_runtime.h>

// Problem constants (from reference): S=4096, B=8, D=H=1024
#define S_LEN 4096
#define B_SZ  8
#define D_IN  1024
#define H_SZ  1024

typedef __attribute__((ext_vector_type(16))) __bf16        v16bf;
typedef __attribute__((ext_vector_type(8)))  float         v8f;
typedef __attribute__((ext_vector_type(4)))  unsigned int  v4u;

union FragU { v4u u2[2]; v16bf bf; };

__device__ __forceinline__ unsigned short f2bf(float x) {
  unsigned int u = __float_as_uint(x);
  unsigned int r = u + 0x7FFFu + ((u >> 16) & 1u);   // round-to-nearest-even
  return (unsigned short)(r >> 16);
}

// CDNA5 async global->LDS copy (ASYNCcnt-tracked, no VGPR staging).
// GV mode: 64-bit global address in VGPR pair, per-lane LDS byte offset in VGPR.
__device__ __forceinline__ void async_copy_b128(unsigned lds_off, const void* gptr) {
  unsigned long long ga = (unsigned long long)gptr;
  asm volatile("global_load_async_to_lds_b128 %0, %1, off"
               :: "v"(lds_off), "v"(ga) : "memory");
}
__device__ __forceinline__ void wait_async_le4() {
  asm volatile("s_wait_asynccnt 0x4" ::: "memory");
}
__device__ __forceinline__ void wait_async_0() {
  asm volatile("s_wait_asynccnt 0x0" ::: "memory");
}

// ---------------------------------------------------------------------------
// Elementwise f32 -> bf16 (vectorized x4)
// ---------------------------------------------------------------------------
__global__ void cvt_f32_bf16(const float* __restrict__ in,
                             unsigned short* __restrict__ out, int n) {
  int i = (blockIdx.x * blockDim.x + threadIdx.x) * 4;
  if (i >= n) return;
  float4 v = *(const float4*)(in + i);
  ushort4 o;
  o.x = f2bf(v.x); o.y = f2bf(v.y); o.z = f2bf(v.z); o.w = f2bf(v.w);
  *(ushort4*)(out + i) = o;
}

// ---------------------------------------------------------------------------
// W [K][N] f32  ->  Wt [N][K] bf16   (32x32 LDS tile transpose)
// ---------------------------------------------------------------------------
__global__ void cvt_transpose(const float* __restrict__ W,
                              unsigned short* __restrict__ Wt, int K, int N) {
  __shared__ unsigned short tile[32][33];
  int nb = blockIdx.x * 32, kb = blockIdx.y * 32;
  int tx = threadIdx.x & 31, ty = threadIdx.x >> 5;  // 32 x 8
#pragma unroll
  for (int i = 0; i < 32; i += 8)
    tile[ty + i][tx] = f2bf(W[(size_t)(kb + ty + i) * N + nb + tx]);
  __syncthreads();
#pragma unroll
  for (int i = 0; i < 32; i += 8)
    Wt[(size_t)(nb + ty + i) * K + kb + tx] = tile[tx][ty + i];
}

// ---------------------------------------------------------------------------
// GEMM: C[M][N] = A[M][K](bf16) * Bt[N][K](bf16)^T + bias, fused sigmoid on
// columns n >= Hthr. fp32 accumulate via v_wmma_f32_16x16x32_bf16.
// Block = 256 threads (8 waves), tile 128x128, BK=32, double-buffered LDS
// fed by GLOBAL_LOAD_ASYNC_TO_LDS_B128 (ASYNCcnt pipeline).
// Wave grid 4(M) x 2(N); each wave: 2x4 16x16 accumulators.
// ---------------------------------------------------------------------------
#define BM 128
#define BN 128
#define BK 32

__global__ __launch_bounds__(256) void gemm_bf16_wmma(
    const unsigned short* __restrict__ A,   // [M][K] bf16 row-major
    const unsigned short* __restrict__ Bt,  // [N][K] bf16 row-major (W^T)
    const float* __restrict__ bias,         // [N]
    float* __restrict__ C,                  // [M][N] f32
    int K, int N, int Hthr) {
  __shared__ __align__(16) unsigned short lsA[2][BM * BK];
  __shared__ __align__(16) unsigned short lsB[2][BN * BK];

  const int tid  = threadIdx.x;
  const int lane = tid & 31;
  const int wave = tid >> 5;
  const int wm   = wave & 3;    // wave M index: 0..3 (rows wm*32..+32)
  const int wn   = wave >> 2;   // wave N index: 0..1 (cols wn*64..+64)
  const int half = lane >> 4;   // lane group (ISA K-half selector)
  const int lr   = lane & 15;

  const int gm0 = blockIdx.y * BM;
  const int gn0 = blockIdx.x * BN;

  // tile loaders: 256 threads cover 128 rows x 2 halves, 32B (2 x b128) each
  const int row  = tid >> 1;
  const int part = tid & 1;
  const unsigned short* aRow = A  + (size_t)(gm0 + row) * K + part * 16;
  const unsigned short* bRow = Bt + (size_t)(gn0 + row) * K + part * 16;

  // LDS byte offsets for async destinations (generic addr low 32 bits == LDS offset)
  unsigned lAoff[2], lBoff[2];
#pragma unroll
  for (int b = 0; b < 2; b++) {
    lAoff[b] = (unsigned)(size_t)&lsA[b][row * BK + part * 16];
    lBoff[b] = (unsigned)(size_t)&lsB[b][row * BK + part * 16];
  }

  v8f acc[2][4];
#pragma unroll
  for (int i = 0; i < 2; i++)
#pragma unroll
    for (int j = 0; j < 4; j++) {
      v8f z = {};
      acc[i][j] = z;
    }

  const int T = K / BK;

  // prologue: stage tile 0 into buffer 0 (4 async b128 per thread per tile)
  async_copy_b128(lAoff[0],      aRow);
  async_copy_b128(lAoff[0] + 16, aRow + 8);
  async_copy_b128(lBoff[0],      bRow);
  async_copy_b128(lBoff[0] + 16, bRow + 8);

  for (int t = 0; t < T; ++t) {
    const int buf = t & 1;
    if (t + 1 < T) {
      // prefetch tile t+1 into the other buffer (its readers passed the
      // trailing barrier of iteration t-1)
      const int nk = (t + 1) * BK;
      async_copy_b128(lAoff[buf ^ 1],      aRow + nk);
      async_copy_b128(lAoff[buf ^ 1] + 16, aRow + nk + 8);
      async_copy_b128(lBoff[buf ^ 1],      bRow + nk);
      async_copy_b128(lBoff[buf ^ 1] + 16, bRow + nk + 8);
      wait_async_le4();   // in-order completion: tile t's 4 ops have landed
    } else {
      wait_async_0();
    }
    __syncthreads();      // all waves' async writes for tile t visible

    const unsigned short* aT = lsA[buf];
    const unsigned short* bT = lsB[buf];

    // A fragments: 16x32 tile, lane half=0 -> K {0..7,16..23}, half=1 -> +8
    FragU af[2];
#pragma unroll
    for (int i = 0; i < 2; i++) {
      int m = wm * 32 + i * 16 + lr;
      af[i].u2[0] = *(const v4u*)&aT[m * BK + 8 * half];
      af[i].u2[1] = *(const v4u*)&aT[m * BK + 16 + 8 * half];
    }
    // B fragments: 32x16 tile; lane column n=lr, half=0 -> K {0..7,16..23}, half=1 -> +8
    FragU bfrag[4];
#pragma unroll
    for (int j = 0; j < 4; j++) {
      int n = wn * 64 + j * 16 + lr;
      bfrag[j].u2[0] = *(const v4u*)&bT[n * BK + 8 * half];
      bfrag[j].u2[1] = *(const v4u*)&bT[n * BK + 16 + 8 * half];
    }
#pragma unroll
    for (int i = 0; i < 2; i++)
#pragma unroll
      for (int j = 0; j < 4; j++)
        acc[i][j] = __builtin_amdgcn_wmma_f32_16x16x32_bf16(
            false, af[i].bf, false, bfrag[j].bf, (short)0, acc[i][j],
            false, false);

    __syncthreads();      // done reading buf before tile t+2 overwrites it
  }

  // Epilogue: C/D layout -> VGPR r holds row (half*8 + r), col = lane%16.
  // Fused: + bias[n]; sigmoid for n >= Hthr (f,o gate columns).
  // Streaming 384MB intermediate: non-temporal stores (2x L2 capacity).
#pragma unroll
  for (int i = 0; i < 2; i++) {
    int mb = gm0 + wm * 32 + i * 16 + half * 8;
#pragma unroll
    for (int j = 0; j < 4; j++) {
      int n = gn0 + wn * 64 + j * 16 + lr;
      float bb = bias[n];
      bool sig = (n >= Hthr);
      v8f a = acc[i][j];
#pragma unroll
      for (int r = 0; r < 8; r++) {
        float v = a[r] + bb;
        if (sig) v = 1.0f / (1.0f + __expf(-v));
        __builtin_nontemporal_store(v, &C[(size_t)(mb + r) * N + n]);
      }
    }
  }
}

// ---------------------------------------------------------------------------
// Fused SRU scan: one thread per (b,h) channel, sequential over S.
//   c = f*c + (1-f)*u ; h = o*tanh(c) + (1-o)*x
// f,o arrive pre-sigmoided from the GEMM epilogue. Also emits bf16 copy of h
// (layer-2 GEMM input) when hbf != nullptr, and the final cell state.
// ---------------------------------------------------------------------------
__global__ void sru_scan(const float* __restrict__ gates,   // [S][B][3H]
                         const float* __restrict__ xin,     // [S][B][H]
                         const float* __restrict__ c0,      // [B*H]
                         float* __restrict__ hout,          // [S][B][H]
                         unsigned short* __restrict__ hbf,  // [S][B][H] or null
                         float* __restrict__ clast,         // [B*H]
                         int S, int B, int H) {
  int idx = blockIdx.x * blockDim.x + threadIdx.x;
  if (idx >= B * H) return;
  int b = idx / H;
  int h = idx - b * H;
  float c = c0[idx];
  const int N3 = 3 * H;
  for (int s = 0; s < S; ++s) {
    size_t base = (size_t)s * B + b;
    size_t gi = base * N3 + h;
    float u = __builtin_nontemporal_load(&gates[gi]);
    float f = __builtin_nontemporal_load(&gates[gi + H]);
    float o = __builtin_nontemporal_load(&gates[gi + 2 * H]);
    c = f * c + (1.0f - f) * u;
    float xv = xin[base * H + h];
    float hv = o * tanhf(c) + (1.0f - o) * xv;
    size_t hi = base * H + h;
    hout[hi] = hv;
    if (hbf) hbf[hi] = f2bf(hv);
  }
  clast[idx] = c;
}

// ---------------------------------------------------------------------------
// Launch: convert -> gemm1 -> scan1 -> (transpose W2) -> gemm2 -> scan2
// ---------------------------------------------------------------------------
extern "C" void kernel_launch(void* const* d_in, const int* in_sizes, int n_in,
                              void* d_out, int out_size, void* d_ws, size_t ws_size,
                              hipStream_t stream) {
  const float* x  = (const float*)d_in[0];
  const float* h0 = (const float*)d_in[1];
  const float* W1 = (const float*)d_in[2];
  const float* b1 = (const float*)d_in[3];
  const float* W2 = (const float*)d_in[4];
  const float* b2 = (const float*)d_in[5];
  float* out = (float*)d_out;

  const int S = S_LEN, B = B_SZ, H = H_SZ;
  const int M  = S * B;    // 32768 GEMM rows
  const int N3 = 3 * H;    // 3072
  const int K  = D_IN;     // 1024

  // workspace carve-up (~646 MB)
  char* p = (char*)d_ws;
  unsigned short* xa  = (unsigned short*)p; p += (size_t)M * K * 2;   // x bf16
  unsigned short* h1b = (unsigned short*)p; p += (size_t)M * H * 2;   // h1 bf16
  unsigned short* Wt  = (unsigned short*)p; p += (size_t)N3 * K * 2;  // W^T bf16 (reused)
  float* gates = (float*)p;                 p += (size_t)M * N3 * 4;  // gates f32 (reused)
  float* h1f   = (float*)p;                 p += (size_t)M * H * 4;   // h1 f32

  // output layout: h2 [S,B,H] then hidden [2,B,H] = (c1_last, c2_last)
  float* h2_out = out;
  float* c1_out = out + (size_t)M * H;
  float* c2_out = c1_out + (size_t)B * H;

  dim3 blk(256);

  // ---- layer 1 ----
  cvt_f32_bf16<<<dim3((unsigned)((size_t)M * K / 1024)), blk, 0, stream>>>(x, xa, M * K);
  cvt_transpose<<<dim3(N3 / 32, K / 32), blk, 0, stream>>>(W1, Wt, K, N3);
  gemm_bf16_wmma<<<dim3(N3 / BN, M / BM), blk, 0, stream>>>(xa, Wt, b1, gates, K, N3, H);
  sru_scan<<<dim3((B * H) / 256), blk, 0, stream>>>(gates, x, h0, h1f, h1b, c1_out, S, B, H);

  // ---- layer 2 ----
  cvt_transpose<<<dim3(N3 / 32, H / 32), blk, 0, stream>>>(W2, Wt, H, N3);
  gemm_bf16_wmma<<<dim3(N3 / BN, M / BM), blk, 0, stream>>>(h1b, Wt, b2, gates, H, N3, H);
  sru_scan<<<dim3((B * H) / 256), blk, 0, stream>>>(gates, h1f, h0 + (size_t)B * H,
                                                    h2_out, nullptr, c2_out, S, B, H);
}
